// TransformerBlockQuantum_65481071396096
// MI455X (gfx1250) — matrix operand order repaired and checked by reading, not compiled, not run on hardware
//
#include <hip/hip_runtime.h>
#include <hip/hip_bf16.h>

// ---------------------------------------------------------------------------
// Types for CDNA5 WMMA (wave32, 16x16x32 bf16 -> f32)
// ---------------------------------------------------------------------------
typedef __bf16 bf16;
typedef bf16  v8bf  __attribute__((ext_vector_type(8)));
typedef bf16  v16bf __attribute__((ext_vector_type(16)));
typedef float v8f   __attribute__((ext_vector_type(8)));
typedef int   v4i   __attribute__((ext_vector_type(4)));

#define EMB   1024
#define SEQ   2048
#define BATCH 4
#define HEADS 16
#define HD    64
#define FFN   4096
#define ROWS  (BATCH * SEQ)   // 8192
#define LN_EPS 1e-5f

// Async global->LDS staging (CDNA5 GLOBAL_LOAD_ASYNC_TO_LDS_B128, ASYNCcnt)
#if defined(__has_builtin)
#if __has_builtin(__builtin_amdgcn_global_load_async_to_lds_b128)
#define HAVE_ASYNC_LDS 1
#endif
#endif

#ifdef HAVE_ASYNC_LDS
// param 1 per clang diagnostic: 'int __vector(4) __device__ *'  (addrspace 1)
// param 2: LDS destination (addrspace 3)
typedef __attribute__((address_space(1))) v4i* gv4i_p;
typedef __attribute__((address_space(3))) v4i* lv4i_p;
static __device__ __forceinline__ void async_copy16(const void* g, void* l) {
    __builtin_amdgcn_global_load_async_to_lds_b128(
        (gv4i_p)(const_cast<void*>(g)), (lv4i_p)l, 0, 0);
}
static __device__ __forceinline__ void async_wait0() {
#if __has_builtin(__builtin_amdgcn_s_wait_asynccnt)
    __builtin_amdgcn_s_wait_asynccnt(0);
#else
    asm volatile("s_wait_asynccnt 0x0" ::: "memory");
#endif
}
#endif

static __device__ __forceinline__ bf16 f2bf(float f) {
    return (bf16)f;   // native RTNE conversion on gfx1250
}

static __device__ __forceinline__ v8f vzero8() {
    v8f z;
#pragma unroll
    for (int i = 0; i < 8; ++i) z[i] = 0.0f;
    return z;
}

// A-fragment: lane holds M=lane%16; elems 0..7 = K(16*half..),
// elems 8..15 = K(16*half+8..). Two contiguous 16B chunks.
static __device__ __forceinline__ v16bf frag_a(const bf16* lo, const bf16* hi) {
    v8bf a = *(const v8bf*)lo;
    v8bf b = *(const v8bf*)hi;
    v16bf r;
#pragma unroll
    for (int i = 0; i < 8; ++i) { r[i] = a[i]; r[8 + i] = b[i]; }
    return r;
}

// B-fragment: lane holds N=lane%16; elems 0..15 = K(16*half .. 16*half+15),
// one contiguous 32B chunk (caller passes base + 16*half).
static __device__ __forceinline__ v16bf frag_b(const bf16* p) {
    v8bf a = *(const v8bf*)p;
    v8bf b = *(const v8bf*)(p + 8);
    v16bf r;
#pragma unroll
    for (int i = 0; i < 8; ++i) { r[i] = a[i]; r[8 + i] = b[i]; }
    return r;
}

#define WMMA_BF16(A, B, C) \
    __builtin_amdgcn_wmma_f32_16x16x32_bf16(false, (A), false, (B), (short)0, (C), false, false)

// ---------------------------------------------------------------------------
// fp32 -> bf16 conversion (grid-stride)
// ---------------------------------------------------------------------------
__global__ __launch_bounds__(256) void cvt_f32_bf16_kernel(const float* __restrict__ in,
                                                           bf16* __restrict__ out, int n) {
    for (int i = blockIdx.x * blockDim.x + threadIdx.x; i < n; i += gridDim.x * blockDim.x)
        out[i] = f2bf(in[i]);
}

// ---------------------------------------------------------------------------
// Tiled WMMA GEMM:  C[M,N](f32 or bf16) = A[M,K](bf16) * B[K,N](bf16)
// Block tile 128x128x32, 256 threads = 8 waves; wave tile 64x32 (4x2 WMMA tiles)
// A tile staged with async global->LDS copies when available.
// ---------------------------------------------------------------------------
__global__ __launch_bounds__(256) void gemm_bf16_wmma_kernel(
    const bf16* __restrict__ A, const bf16* __restrict__ B,
    void* __restrict__ Cout, int M, int N, int K, int c_is_bf16) {

    __shared__ __align__(16) bf16 Alds[128][32];   // [m][k]
    __shared__ __align__(16) bf16 Blds[128][32];   // [n][k]  (transposed stage)

    const int tid  = threadIdx.x;
    const int bm   = blockIdx.y * 128;
    const int bn   = blockIdx.x * 128;
    const int wid  = tid >> 5;
    const int lane = tid & 31;
    const int wm   = wid >> 2;          // 0..1
    const int wn   = wid & 3;           // 0..3
    const int half = lane >> 4;
    const int l16  = lane & 15;
    const int klo  = half * 16;
    const int khi  = half * 16 + 8;

    v8f acc[4][2];
#pragma unroll
    for (int i = 0; i < 4; ++i)
#pragma unroll
        for (int j = 0; j < 2; ++j) acc[i][j] = vzero8();

    const int arow = tid >> 1;          // 0..127
    const int acol = (tid & 1) * 16;    // 0 / 16
    const int brow = tid >> 3;          // 0..31
    const int bcol = (tid & 7) * 16;    // 0..112

    for (int kt = 0; kt < K; kt += 32) {
        // stage A (row-major, straight copy -> async path)
        const bf16* ag = A + (size_t)(bm + arow) * K + kt + acol;
#ifdef HAVE_ASYNC_LDS
        async_copy16(ag,     &Alds[arow][acol]);
        async_copy16(ag + 8, &Alds[arow][acol + 8]);
#else
        *(v8bf*)&Alds[arow][acol]     = *(const v8bf*)ag;
        *(v8bf*)&Alds[arow][acol + 8] = *(const v8bf*)(ag + 8);
#endif
        // stage B transposed: Blds[n][k]
        const bf16* bg = B + (size_t)(kt + brow) * N + bn + bcol;
        v8bf b0 = *(const v8bf*)bg;
        v8bf b1 = *(const v8bf*)(bg + 8);
#pragma unroll
        for (int i = 0; i < 8; ++i) {
            Blds[bcol + i][brow]     = b0[i];
            Blds[bcol + 8 + i][brow] = b1[i];
        }
#ifdef HAVE_ASYNC_LDS
        async_wait0();
#endif
        __syncthreads();

        v16bf af[4], bfr[2];
#pragma unroll
        for (int i = 0; i < 4; ++i) {
            int m = wm * 64 + i * 16 + l16;
            af[i] = frag_a(&Alds[m][klo], &Alds[m][khi]);
        }
#pragma unroll
        for (int j = 0; j < 2; ++j) {
            int n = wn * 32 + j * 16 + l16;
            bfr[j] = frag_b(&Blds[n][16 * half]);
        }
#pragma unroll
        for (int i = 0; i < 4; ++i)
#pragma unroll
            for (int j = 0; j < 2; ++j)
                acc[i][j] = WMMA_BF16(af[i], bfr[j], acc[i][j]);
        __syncthreads();
    }

    // epilogue: C layout — VGPR r holds M = 8*half + r, N = l16
#pragma unroll
    for (int i = 0; i < 4; ++i)
#pragma unroll
        for (int j = 0; j < 2; ++j) {
            int m0 = bm + wm * 64 + i * 16 + half * 8;
            int n0 = bn + wn * 32 + j * 16 + l16;
            if (c_is_bf16) {
                bf16* C = (bf16*)Cout;
#pragma unroll
                for (int r = 0; r < 8; ++r)
                    C[(size_t)(m0 + r) * N + n0] = f2bf(acc[i][j][r]);
            } else {
                float* C = (float*)Cout;
#pragma unroll
                for (int r = 0; r < 8; ++r)
                    C[(size_t)(m0 + r) * N + n0] = acc[i][j][r];
            }
        }
}

// ---------------------------------------------------------------------------
// Flash-attention (bf16 WMMA, online softmax).
// grid = (B*H, S/64), block = 128 (4 waves); each wave owns 16 query rows.
// Q/K/V stored as [B,S,E] with head h in columns h*64..h*64+63.
// ---------------------------------------------------------------------------
__global__ __launch_bounds__(128) void attn_wmma_kernel(
    const bf16* __restrict__ Qp, const bf16* __restrict__ Kp,
    const bf16* __restrict__ Vp, bf16* __restrict__ ctx) {

    const int bh = blockIdx.x;          // 0..63
    const int b  = bh >> 4;
    const int h  = bh & 15;
    const int tid  = threadIdx.x;
    const int wid  = tid >> 5;
    const int lane = tid & 31;
    const int half = lane >> 4;
    const int l16  = lane & 15;
    const int klo  = half * 16;
    const int khi  = half * 16 + 8;
    const int q0   = blockIdx.y * 64 + wid * 16;
    const size_t base = ((size_t)b * SEQ) * EMB + (size_t)h * HD;

    __shared__ __align__(16) bf16 Vlds[64][32];      // [d][t_local]
    __shared__ __align__(16) bf16 Plds[4][16][32];   // per-wave P tile

    // Q fragments (Dk=64 -> two K=32 frags), loaded once, direct from global
    v16bf qa[2];
    {
        const bf16* qr = Qp + base + (size_t)(q0 + l16) * EMB;
        qa[0] = frag_a(qr + klo, qr + khi);
        qa[1] = frag_a(qr + 32 + klo, qr + 32 + khi);
    }

    float mrun[8], lrun[8];
    v8f cacc[4];
#pragma unroll
    for (int j = 0; j < 8; ++j) { mrun[j] = -3.0e38f; lrun[j] = 0.0f; }
#pragma unroll
    for (int dt = 0; dt < 4; ++dt) cacc[dt] = vzero8();

    const int trow  = tid >> 2;          // 0..31
    const int dbase = (tid & 3) * 16;    // 0..48

    for (int t0 = 0; t0 < SEQ; t0 += 32) {
        // stage V chunk transposed into LDS: Vlds[d][t]
        const bf16* vg = Vp + base + (size_t)(t0 + trow) * EMB + dbase;
        v8bf v0 = *(const v8bf*)vg;
        v8bf v1 = *(const v8bf*)(vg + 8);
#pragma unroll
        for (int i = 0; i < 8; ++i) {
            Vlds[dbase + i][trow]     = v0[i];
            Vlds[dbase + 8 + i][trow] = v1[i];
        }
        __syncthreads();

        // scores: S[16x32] = Q(16x64) * K^T(64x32), two 16-col tiles
        v8f s[2];
#pragma unroll
        for (int jt = 0; jt < 2; ++jt) {
            const bf16* kr = Kp + base + (size_t)(t0 + jt * 16 + l16) * EMB;
            v16bf kb0 = frag_b(kr + 16 * half);          // d = 0..31
            v16bf kb1 = frag_b(kr + 32 + 16 * half);     // d = 32..63
            v8f z = vzero8();
            z = WMMA_BF16(qa[0], kb0, z);
            z = WMMA_BF16(qa[1], kb1, z);
            s[jt] = z * 0.125f;                          // 1/sqrt(64)
        }

        // online softmax per row (row m = 8*half + j, reduced over 16-lane half)
#pragma unroll
        for (int j = 0; j < 8; ++j) {
            float mloc = fmaxf(s[0][j], s[1][j]);
#pragma unroll
            for (int mask = 1; mask < 16; mask <<= 1)
                mloc = fmaxf(mloc, __shfl_xor(mloc, mask, 32));
            float mnew  = fmaxf(mrun[j], mloc);
            float alpha = __expf(mrun[j] - mnew);
            float p0 = __expf(s[0][j] - mnew);
            float p1 = __expf(s[1][j] - mnew);
            float rs = p0 + p1;
#pragma unroll
            for (int mask = 1; mask < 16; mask <<= 1)
                rs += __shfl_xor(rs, mask, 32);
            lrun[j] = lrun[j] * alpha + rs;
            mrun[j] = mnew;
#pragma unroll
            for (int dt = 0; dt < 4; ++dt) cacc[dt][j] *= alpha;
            // P tile to LDS (C-layout -> row-major)
            Plds[wid][half * 8 + j][l16]      = f2bf(p0);
            Plds[wid][half * 8 + j][16 + l16] = f2bf(p1);
        }

        // ctx += P(16x32) * V(32x64); P reload in A-layout (same wave, LDS in-order)
        v16bf pa = frag_a(&Plds[wid][l16][klo], &Plds[wid][l16][khi]);
#pragma unroll
        for (int dt = 0; dt < 4; ++dt) {
            v16bf vb = frag_b(&Vlds[dt * 16 + l16][16 * half]);
            cacc[dt] = WMMA_BF16(pa, vb, cacc[dt]);
        }
        __syncthreads();   // before next V overwrite
    }

    // normalize and write ctx (bf16) back in [B,S,E] layout
#pragma unroll
    for (int dt = 0; dt < 4; ++dt)
#pragma unroll
        for (int j = 0; j < 8; ++j) {
            int row = q0 + half * 8 + j;
            int col = dt * 16 + l16;
            ctx[base + (size_t)row * EMB + col] = f2bf(cacc[dt][j] / lrun[j]);
        }
}

// ---------------------------------------------------------------------------
// Fused: x1 = LN1(x + attn_out); q = cos(2*x1[:8]+theta); h = relu(q@w1) (bf16)
// One block per row (256 threads).
// ---------------------------------------------------------------------------
__global__ __launch_bounds__(256) void ln1_quantum_kernel(
    const float* __restrict__ x, const float* __restrict__ attn,
    const float* __restrict__ g, const float* __restrict__ bt,
    const float* __restrict__ theta, const float* __restrict__ w1,
    float* __restrict__ x1f, bf16* __restrict__ hb) {

    const int row = blockIdx.x;
    const int tid = threadIdx.x;
    const size_t rb = (size_t)row * EMB;

    float s = 0.f, s2 = 0.f;
    for (int c = tid; c < EMB; c += 256) {
        float v = x[rb + c] + attn[rb + c];
        s += v; s2 += v * v;
    }
#pragma unroll
    for (int mask = 1; mask < 32; mask <<= 1) {
        s  += __shfl_xor(s,  mask, 32);
        s2 += __shfl_xor(s2, mask, 32);
    }
    __shared__ float shs[8], shs2[8], stats[2];
    if ((tid & 31) == 0) { shs[tid >> 5] = s; shs2[tid >> 5] = s2; }
    __syncthreads();
    if (tid == 0) {
        float ts = 0.f, ts2 = 0.f;
#pragma unroll
        for (int i = 0; i < 8; ++i) { ts += shs[i]; ts2 += shs2[i]; }
        float mean = ts / EMB;
        float var  = ts2 / EMB - mean * mean;
        stats[0] = mean;
        stats[1] = rsqrtf(var + LN_EPS);
    }
    __syncthreads();
    const float mean = stats[0], rstd = stats[1];

    for (int c = tid; c < EMB; c += 256) {
        float v = x[rb + c] + attn[rb + c];
        x1f[rb + c] = (v - mean) * rstd * g[c] + bt[c];
    }

    __shared__ float qs[8];
    if (tid < 8) {
        float v  = x[rb + tid] + attn[rb + tid];
        float x1 = (v - mean) * rstd * g[tid] + bt[tid];
        qs[tid] = __cosf(2.0f * x1 + theta[tid]);
    }
    __syncthreads();

    float q[8];
#pragma unroll
    for (int i = 0; i < 8; ++i) q[i] = qs[i];
    for (int f = tid; f < FFN; f += 256) {
        float a = 0.f;
#pragma unroll
        for (int i = 0; i < 8; ++i) a += q[i] * w1[i * FFN + f];
        hb[(size_t)row * FFN + f] = f2bf(fmaxf(a, 0.f));
    }
}

// ---------------------------------------------------------------------------
// out = LN2(x1 + ffn_out)
// ---------------------------------------------------------------------------
__global__ __launch_bounds__(256) void ln2_kernel(
    const float* __restrict__ x1f, const float* __restrict__ ffn,
    const float* __restrict__ g, const float* __restrict__ bt,
    float* __restrict__ out) {

    const int row = blockIdx.x;
    const int tid = threadIdx.x;
    const size_t rb = (size_t)row * EMB;

    float s = 0.f, s2 = 0.f;
    for (int c = tid; c < EMB; c += 256) {
        float v = x1f[rb + c] + ffn[rb + c];
        s += v; s2 += v * v;
    }
#pragma unroll
    for (int mask = 1; mask < 32; mask <<= 1) {
        s  += __shfl_xor(s,  mask, 32);
        s2 += __shfl_xor(s2, mask, 32);
    }
    __shared__ float shs[8], shs2[8], stats[2];
    if ((tid & 31) == 0) { shs[tid >> 5] = s; shs2[tid >> 5] = s2; }
    __syncthreads();
    if (tid == 0) {
        float ts = 0.f, ts2 = 0.f;
#pragma unroll
        for (int i = 0; i < 8; ++i) { ts += shs[i]; ts2 += shs2[i]; }
        float mean = ts / EMB;
        float var  = ts2 / EMB - mean * mean;
        stats[0] = mean;
        stats[1] = rsqrtf(var + LN_EPS);
    }
    __syncthreads();
    const float mean = stats[0], rstd = stats[1];

    for (int c = tid; c < EMB; c += 256) {
        float v = x1f[rb + c] + ffn[rb + c];
        out[rb + c] = (v - mean) * rstd * g[c] + bt[c];
    }
}

// ---------------------------------------------------------------------------
// Host launch
// ---------------------------------------------------------------------------
extern "C" void kernel_launch(void* const* d_in, const int* in_sizes, int n_in,
                              void* d_out, int out_size, void* d_ws, size_t ws_size,
                              hipStream_t stream) {
    (void)in_sizes; (void)n_in; (void)out_size; (void)ws_size;

    const float* x     = (const float*)d_in[0];
    const float* wq    = (const float*)d_in[1];
    const float* wk    = (const float*)d_in[2];
    const float* wv    = (const float*)d_in[3];
    const float* wo    = (const float*)d_in[4];
    const float* theta = (const float*)d_in[5];
    const float* w1    = (const float*)d_in[6];
    const float* w2    = (const float*)d_in[7];
    const float* ln1g  = (const float*)d_in[8];
    const float* ln1b  = (const float*)d_in[9];
    const float* ln2g  = (const float*)d_in[10];
    const float* ln2b  = (const float*)d_in[11];
    float* out = (float*)d_out;

    char* ws = (char*)d_ws;
    size_t off = 0;
    auto alloc = [&](size_t bytes) { char* p = ws + off; off += (bytes + 255) & ~size_t(255); return p; };

    bf16*  xb    = (bf16*)alloc((size_t)ROWS * EMB * 2);
    bf16*  wqb   = (bf16*)alloc((size_t)EMB * EMB * 2);
    bf16*  wkb   = (bf16*)alloc((size_t)EMB * EMB * 2);
    bf16*  wvb   = (bf16*)alloc((size_t)EMB * EMB * 2);
    bf16*  wob   = (bf16*)alloc((size_t)EMB * EMB * 2);
    bf16*  w2b   = (bf16*)alloc((size_t)FFN * EMB * 2);
    bf16*  Qb    = (bf16*)alloc((size_t)ROWS * EMB * 2);
    bf16*  Kb    = (bf16*)alloc((size_t)ROWS * EMB * 2);
    bf16*  Vb    = (bf16*)alloc((size_t)ROWS * EMB * 2);
    bf16*  ctxb  = (bf16*)alloc((size_t)ROWS * EMB * 2);
    float* attnf = (float*)alloc((size_t)ROWS * EMB * 4);
    float* x1f   = (float*)alloc((size_t)ROWS * EMB * 4);
    bf16*  hb    = (bf16*)alloc((size_t)ROWS * FFN * 2);
    float* ffnf  = (float*)alloc((size_t)ROWS * EMB * 4);

    // 1. fp32 -> bf16 conversions
    cvt_f32_bf16_kernel<<<4096, 256, 0, stream>>>(x,  xb,  ROWS * EMB);
    cvt_f32_bf16_kernel<<<1024, 256, 0, stream>>>(wq, wqb, EMB * EMB);
    cvt_f32_bf16_kernel<<<1024, 256, 0, stream>>>(wk, wkb, EMB * EMB);
    cvt_f32_bf16_kernel<<<1024, 256, 0, stream>>>(wv, wvb, EMB * EMB);
    cvt_f32_bf16_kernel<<<1024, 256, 0, stream>>>(wo, wob, EMB * EMB);
    cvt_f32_bf16_kernel<<<4096, 256, 0, stream>>>(w2, w2b, FFN * EMB);

    // 2. QKV projections (bf16 out)
    dim3 ggrid(EMB / 128, ROWS / 128);
    gemm_bf16_wmma_kernel<<<ggrid, 256, 0, stream>>>(xb, wqb, Qb, ROWS, EMB, EMB, 1);
    gemm_bf16_wmma_kernel<<<ggrid, 256, 0, stream>>>(xb, wkb, Kb, ROWS, EMB, EMB, 1);
    gemm_bf16_wmma_kernel<<<ggrid, 256, 0, stream>>>(xb, wvb, Vb, ROWS, EMB, EMB, 1);

    // 3. flash attention
    attn_wmma_kernel<<<dim3(BATCH * HEADS, SEQ / 64), 128, 0, stream>>>(Qb, Kb, Vb, ctxb);

    // 4. output projection (f32 out)
    gemm_bf16_wmma_kernel<<<ggrid, 256, 0, stream>>>(ctxb, wob, attnf, ROWS, EMB, EMB, 0);

    // 5. residual + LN1 + quantum FFN first stage
    ln1_quantum_kernel<<<ROWS, 256, 0, stream>>>(x, attnf, ln1g, ln1b, theta, w1, x1f, hb);

    // 6. h @ w2 (f32 out)
    gemm_bf16_wmma_kernel<<<ggrid, 256, 0, stream>>>(hb, w2b, ffnf, ROWS, EMB, FFN, 0);

    // 7. residual + LN2 -> output
    ln2_kernel<<<ROWS, 256, 0, stream>>>(x1f, ffnf, ln2g, ln2b, out);
}
